// CrossViewSwapAttention_29807073034953
// MI455X (gfx1250) — compile-verified, matrix operand order checked by reading, not run on hardware
//
#include <hip/hip_runtime.h>

typedef __attribute__((ext_vector_type(16))) _Float16 v16h;
typedef __attribute__((ext_vector_type(8)))  float    v8f;

#define HEADS 4
#define DH    32
#define DDIM  128
#define XDIM  8
#define YDIM  8
#define WW1   16
#define WW2   16
#define KW    6
#define LWIN  64
#define KN    216
#define KNP   224
#define SCALE 0.17677669529663687f  // 32^-0.5

__device__ __forceinline__ int frag_k(int lane, int e) {
    // f16 A/B fragment K-index mapping (ISA 7.12.2, 16-bit A 16x32)
    return (e & 7) | ((lane >> 4) << 3) | ((e >> 3) << 4);
}

__device__ __forceinline__ v8f wmma_f16(v16h a, v16h b, v8f c) {
    return __builtin_amdgcn_wmma_f32_16x16x32_f16(false, a, false, b, (short)0, c,
                                                  false, false);
}

// 16-byte copy helper (both sides 16B aligned by construction)
__device__ __forceinline__ void cp16(_Float16* dst, const _Float16* src) {
    *(uint4*)dst = *(const uint4*)src;
}

union h8u4 { _Float16 h[8]; uint4 u; };

// ---------------------------------------------------------------------------
// Kernel 1: convert four 128x128 f32 weights to f16 TRANSPOSED ([out][in])
// so WMMA B-fragments read contiguous k-runs from LDS.
// layout: [WqT | WkT | WvT | WpT], each 16384 halves
// ---------------------------------------------------------------------------
__global__ void convert_w_kernel(const float* __restrict__ wq,
                                 const float* __restrict__ wk,
                                 const float* __restrict__ wv,
                                 const float* __restrict__ wp,
                                 _Float16* __restrict__ w16) {
    int i = blockIdx.x * blockDim.x + threadIdx.x;
    if (i < 16384) {
        int r = i >> 7, c = i & 127;       // r = in(k), c = out(n)
        int t = c * 128 + r;               // transposed index [n][k]
        w16[t]         = (_Float16)wq[i];
        w16[16384 + t] = (_Float16)wk[i];
        w16[32768 + t] = (_Float16)wv[i];
        w16[49152 + t] = (_Float16)wp[i];
    }
}

// ---------------------------------------------------------------------------
// Kernel 2: out = skip (atomicAdd of z/6 contributions comes later)
// ---------------------------------------------------------------------------
__global__ void init_out_kernel(const float* __restrict__ skip,
                                float* __restrict__ out, int n) {
    int i = blockIdx.x * blockDim.x + threadIdx.x;
    if (i < n) out[i] = skip[i];
}

// ---------------------------------------------------------------------------
// Kernel 3: per-window K/V layernorm + projection.
//   kh  -> row-major   [64][224][128]  (scores B-frag wants [key][d])
//   vhT -> transposed  [64][128][224]  (AV B-frag wants [d][key]); the C-tile
//          column-per-lane layout makes this a single b128 store per tile.
// one block per window, 256 threads (8 waves). Weights staged in LDS.
// ---------------------------------------------------------------------------
__global__ void kv_proj_kernel(const float* __restrict__ kin,
                               const float* __restrict__ vin,
                               const float* __restrict__ g_k, const float* __restrict__ b_k,
                               const float* __restrict__ g_v, const float* __restrict__ b_v,
                               const float* __restrict__ bias_k, const float* __restrict__ bias_v,
                               const _Float16* __restrict__ wk16t,
                               const _Float16* __restrict__ wv16t,
                               _Float16* __restrict__ kh, _Float16* __restrict__ vhT) {
    extern __shared__ char smem_raw[];
    _Float16* lnbuf = (_Float16*)smem_raw;      // [KNP][DDIM] = 57344 B
    _Float16* wbufT = lnbuf + KNP * DDIM;       // [128][128]  = 32768 B  ([n][k])

    const int l = blockIdx.x;
    const int x = l >> 3, y = l & 7;
    const int tid = threadIdx.x;
    const int wave = tid >> 5, lane = tid & 31;

    for (int pass = 0; pass < 2; ++pass) {
        const float* src     = pass ? vin : kin;
        const float* g       = pass ? g_v : g_k;
        const float* b       = pass ? b_v : b_k;
        const float* bias    = pass ? bias_v : bias_k;
        const _Float16* w    = pass ? wv16t : wk16t;
        _Float16* dstK       = kh  + (size_t)l * KNP * DDIM;
        _Float16* dstVT      = vhT + (size_t)l * DDIM * KNP;

        // --- layernorm rows into LDS (pad rows KN..KNP-1 with zeros) ---
        if (tid < KNP) {
            if (tid < KN) {
                int n = tid / 36, r = tid % 36;
                int r1 = r / 6, r2 = r % 6;
                const float* row =
                    src + (((((size_t)n * XDIM + x) * YDIM + y) * KW + r1) * KW + r2) * DDIM;
                float mu = 0.f;
                for (int c = 0; c < DDIM; ++c) mu += row[c];
                mu *= (1.0f / DDIM);
                float var = 0.f;
                for (int c = 0; c < DDIM; ++c) { float t = row[c] - mu; var += t * t; }
                var *= (1.0f / DDIM);
                float rs = rsqrtf(var + 1e-5f);
                for (int c = 0; c < DDIM; ++c)
                    lnbuf[tid * DDIM + c] = (_Float16)((row[c] - mu) * rs * g[c] + b[c]);
            } else {
                for (int c = 0; c < DDIM; ++c) lnbuf[tid * DDIM + c] = (_Float16)0.f;
            }
        }
        // --- stage transposed weight matrix into LDS (16B chunks) ---
        for (int c = tid; c < 2048; c += 256) cp16(wbufT + c * 8, w + c * 8);
        __syncthreads();

        // --- GEMM [224x128] = lnbuf x W; wave = N-tile, loop M-tiles ---
        const int nt = wave;
        for (int mt = 0; mt < 14; ++mt) {
            v8f acc = {};
            #pragma unroll
            for (int kk = 0; kk < 4; ++kk) {
                v16h a, bf;
                int mn = lane & 15;
                #pragma unroll
                for (int e = 0; e < 16; ++e) {
                    int ko = frag_k(lane, e);
                    a[e]  = lnbuf[(mt * 16 + mn) * DDIM + kk * 32 + ko];
                    bf[e] = wbufT[(nt * 16 + mn) * DDIM + kk * 32 + ko];
                }
                acc = wmma_f16(a, bf, acc);
            }
            int col = nt * 16 + (lane & 15);
            float bval = bias[col];
            if (pass == 0) {
                #pragma unroll
                for (int i = 0; i < 8; ++i) {
                    int rowi = mt * 16 + i + ((lane >> 4) << 3);
                    dstK[rowi * DDIM + col] = (_Float16)(acc[i] + bval);
                }
            } else {
                h8u4 t;
                #pragma unroll
                for (int i = 0; i < 8; ++i) t.h[i] = (_Float16)(acc[i] + bval);
                int row0 = mt * 16 + ((lane >> 4) << 3);
                *(uint4*)(dstVT + (size_t)col * KNP + row0) = t.u;
            }
        }
        __syncthreads();
    }
}

// ---------------------------------------------------------------------------
// Kernel 4: attention. grid (12 q-tiles, 64 windows), 256 threads (8 waves).
// All WMMA operands fed from LDS in [n][k]-contiguous layouts.
// ---------------------------------------------------------------------------
__global__ void attn_kernel(const float* __restrict__ qin,
                            const float* __restrict__ gq, const float* __restrict__ bqln,
                            const float* __restrict__ bq, const float* __restrict__ bp,
                            const _Float16* __restrict__ wq16t,
                            const _Float16* __restrict__ wp16t,
                            const _Float16* __restrict__ kh,
                            const _Float16* __restrict__ vhT,
                            float* __restrict__ out) {
    extern __shared__ char smem_raw[];
    _Float16* abuf  = (_Float16*)smem_raw;   // [128][128]  qln, later attn output
    _Float16* qh    = abuf + 128 * 128;      // [128][128]  projected q (f16, scaled)
    _Float16* pbuf  = qh + 128 * 128;        // [128][224]  softmax probs f16
    _Float16* stage = pbuf + 128 * KNP;      // 16384 halves: WqT/WpT OR khh+vhhT
    _Float16* khh   = stage;                 // [224][32]   kh head slice   (7168)
    _Float16* vhhT  = stage + KNP * DH;      // [32][224]   vhT head slice  (7168)

    const int qtile = blockIdx.x;
    const int l     = blockIdx.y;
    const int x = l >> 3, y = l & 7;
    const int tid  = threadIdx.x;
    const int wave = tid >> 5, lane = tid & 31;
    const _Float16* khw  = kh  + (size_t)l * KNP * DDIM;
    const _Float16* vhwT = vhT + (size_t)l * DDIM * KNP;

    // --- 1. layernorm of 128 q rows; concurrently stage WqT into LDS ---
    if (tid < 128) {
        int qn = qtile * 128 + tid;
        int n = qn >> 8, p = qn & 255;
        int w1 = p >> 4, w2 = p & 15;
        const float* row =
            qin + (((((size_t)n * XDIM + x) * YDIM + y) * WW1 + w1) * WW2 + w2) * DDIM;
        float mu = 0.f;
        for (int c = 0; c < DDIM; ++c) mu += row[c];
        mu *= (1.0f / DDIM);
        float var = 0.f;
        for (int c = 0; c < DDIM; ++c) { float t = row[c] - mu; var += t * t; }
        var *= (1.0f / DDIM);
        float rs = rsqrtf(var + 1e-5f);
        for (int c = 0; c < DDIM; ++c)
            abuf[tid * DDIM + c] = (_Float16)((row[c] - mu) * rs * gq[c] + bqln[c]);
    }
    for (int c = tid; c < 2048; c += 256) cp16(stage + c * 8, wq16t + c * 8);
    __syncthreads();

    // --- 2. qh = (qln x Wq + bq) * scale, stored f16 ---
    {
        const int nt = wave;
        for (int mt = 0; mt < 8; ++mt) {
            v8f acc = {};
            #pragma unroll
            for (int kk = 0; kk < 4; ++kk) {
                v16h a, bf;
                int mn = lane & 15;
                #pragma unroll
                for (int e = 0; e < 16; ++e) {
                    int ko = frag_k(lane, e);
                    a[e]  = abuf[(mt * 16 + mn) * DDIM + kk * 32 + ko];
                    bf[e] = stage[(nt * 16 + mn) * DDIM + kk * 32 + ko];
                }
                acc = wmma_f16(a, bf, acc);
            }
            int col = nt * 16 + (lane & 15);
            float bval = bq[col];
            #pragma unroll
            for (int i = 0; i < 8; ++i) {
                int rowi = mt * 16 + i + ((lane >> 4) << 3);
                qh[rowi * DDIM + col] = (_Float16)((acc[i] + bval) * SCALE);
            }
        }
    }
    __syncthreads();

    // --- 3. per-head attention; wave owns q rows [wave*16, wave*16+16) ---
    for (int h = 0; h < HEADS; ++h) {
        // stage kh [224][32] and vhT [32][224] head slices (16B chunks)
        for (int c = tid; c < 1792; c += 256) {
            if (c < 896) {                    // khh: row = key, 4 chunks/row
                int row = c >> 2, sub = c & 3;
                cp16(khh + row * DH + sub * 8, khw + row * DDIM + h * DH + sub * 8);
            } else {                          // vhhT: row = d, 28 chunks/row
                int idx = c - 896;
                int d = idx / 28, sub = idx % 28;
                cp16(vhhT + d * KNP + sub * 8, vhwT + (h * DH + d) * KNP + sub * 8);
            }
        }
        // prefetch next head's slices into cache (global_prefetch_b8)
        if (h < 3) {
            if (tid < KNP) __builtin_prefetch(khw + tid * DDIM + (h + 1) * DH, 0, 1);
            if (tid < DH)  __builtin_prefetch(vhwT + ((h + 1) * DH + tid) * KNP, 0, 1);
        }
        __syncthreads();

        // A fragment: 16 q rows x K=32 (head slice) -- one fragment per head
        v16h aq;
        {
            int mn = lane & 15;
            #pragma unroll
            for (int e = 0; e < 16; ++e)
                aq[e] = qh[(wave * 16 + mn) * DDIM + h * DH + frag_k(lane, e)];
        }
        // scores: 14 key tiles, each a single K=32 wmma; kept in registers
        v8f S[14];
        #pragma unroll
        for (int kt = 0; kt < 14; ++kt) {
            v16h bk;
            int nn = lane & 15;
            #pragma unroll
            for (int e = 0; e < 16; ++e)
                bk[e] = khh[(kt * 16 + nn) * DH + frag_k(lane, e)];
            v8f z = {};
            S[kt] = wmma_f16(aq, bk, z);
            if (kt * 16 + nn >= KN) {           // mask padded keys
                #pragma unroll
                for (int i = 0; i < 8; ++i) S[kt][i] = -1e30f;
            }
        }
        // softmax over 224 keys: per-lane partials then xor-shuffle across the
        // 16-lane half (rows 0-7 live in lanes 0-15, rows 8-15 in lanes 16-31)
        float mx[8], sm[8];
        #pragma unroll
        for (int i = 0; i < 8; ++i) mx[i] = -1e30f;
        #pragma unroll
        for (int kt = 0; kt < 14; ++kt)
            #pragma unroll
            for (int i = 0; i < 8; ++i) mx[i] = fmaxf(mx[i], S[kt][i]);
        #pragma unroll
        for (int off = 1; off < 16; off <<= 1)
            #pragma unroll
            for (int i = 0; i < 8; ++i) mx[i] = fmaxf(mx[i], __shfl_xor(mx[i], off, 32));
        #pragma unroll
        for (int i = 0; i < 8; ++i) sm[i] = 0.f;
        #pragma unroll
        for (int kt = 0; kt < 14; ++kt)
            #pragma unroll
            for (int i = 0; i < 8; ++i) {
                float e_ = __expf(S[kt][i] - mx[i]);
                S[kt][i] = e_;
                sm[i] += e_;
            }
        #pragma unroll
        for (int off = 1; off < 16; off <<= 1)
            #pragma unroll
            for (int i = 0; i < 8; ++i) sm[i] += __shfl_xor(sm[i], off, 32);
        float inv[8];
        #pragma unroll
        for (int i = 0; i < 8; ++i) inv[i] = 1.f / sm[i];

        // write normalized probabilities P to LDS (f16) in row-major [q][key]
        {
            int nn = lane & 15;
            #pragma unroll
            for (int kt = 0; kt < 14; ++kt) {
                int key = kt * 16 + nn;
                #pragma unroll
                for (int i = 0; i < 8; ++i) {
                    int rowi = wave * 16 + i + ((lane >> 4) << 3);
                    pbuf[rowi * KNP + key] = (_Float16)(S[kt][i] * inv[i]);
                }
            }
        }
        __syncthreads();

        // AV: O(16 x 32) = P(16 x 224) x vh_head(224 x 32); 7 K-chunks, 2 N-tiles
        v8f o0 = {}, o1 = {};
        #pragma unroll
        for (int kc = 0; kc < 7; ++kc) {
            v16h ap, bv0, bv1;
            int mn = lane & 15;
            #pragma unroll
            for (int e = 0; e < 16; ++e) {
                int ko = frag_k(lane, e);
                ap[e]  = pbuf[(wave * 16 + mn) * KNP + kc * 32 + ko];
                bv0[e] = vhhT[mn * KNP + kc * 32 + ko];
                bv1[e] = vhhT[(16 + mn) * KNP + kc * 32 + ko];
            }
            o0 = wmma_f16(ap, bv0, o0);
            o1 = wmma_f16(ap, bv1, o1);
        }
        // store head output into abuf (reusing the qln buffer)
        {
            int nn = lane & 15;
            #pragma unroll
            for (int i = 0; i < 8; ++i) {
                int rowi = wave * 16 + i + ((lane >> 4) << 3);
                abuf[rowi * DDIM + h * DH + nn]      = (_Float16)o0[i];
                abuf[rowi * DDIM + h * DH + 16 + nn] = (_Float16)o1[i];
            }
        }
        __syncthreads();
    }

    // --- 4. z = a x Wp + bp; out += z / 6 (mean over views via atomics) ---
    for (int c = tid; c < 2048; c += 256) cp16(stage + c * 8, wp16t + c * 8);
    __syncthreads();
    {
        const int nt = wave;
        for (int mt = 0; mt < 8; ++mt) {
            v8f acc = {};
            #pragma unroll
            for (int kk = 0; kk < 4; ++kk) {
                v16h a, bf;
                int mn = lane & 15;
                #pragma unroll
                for (int e = 0; e < 16; ++e) {
                    int ko = frag_k(lane, e);
                    a[e]  = abuf[(mt * 16 + mn) * DDIM + kk * 32 + ko];
                    bf[e] = stage[(nt * 16 + mn) * DDIM + kk * 32 + ko];
                }
                acc = wmma_f16(a, bf, acc);
            }
            int col = nt * 16 + (lane & 15);
            float bval = bp[col];
            #pragma unroll
            for (int i = 0; i < 8; ++i) {
                int rowi = mt * 16 + i + ((lane >> 4) << 3);
                int qn = qtile * 128 + rowi;
                int p = qn & 255;  // (w1,w2) position; view index folds via atomics
                float val = (acc[i] + bval) * (1.0f / 6.0f);
                atomicAdd(&out[((size_t)l * 256 + p) * DDIM + col], val);
            }
        }
    }
}

// ---------------------------------------------------------------------------
extern "C" void kernel_launch(void* const* d_in, const int* in_sizes, int n_in,
                              void* d_out, int out_size, void* d_ws, size_t ws_size,
                              hipStream_t stream) {
    (void)in_sizes; (void)n_in; (void)ws_size;
    const float* q    = (const float*)d_in[0];
    const float* k    = (const float*)d_in[1];
    const float* v    = (const float*)d_in[2];
    const float* skip = (const float*)d_in[3];
    const float* gq   = (const float*)d_in[4];
    const float* bqln = (const float*)d_in[5];
    const float* gk   = (const float*)d_in[6];
    const float* bkln = (const float*)d_in[7];
    const float* gv   = (const float*)d_in[8];
    const float* bvln = (const float*)d_in[9];
    const float* Wq   = (const float*)d_in[10];
    const float* bq   = (const float*)d_in[11];
    const float* Wk   = (const float*)d_in[12];
    const float* bk   = (const float*)d_in[13];
    const float* Wv   = (const float*)d_in[14];
    const float* bv   = (const float*)d_in[15];
    const float* Wp   = (const float*)d_in[16];
    const float* bp   = (const float*)d_in[17];
    float* out = (float*)d_out;

    // workspace carve-up (f16): transposed weights (4*16384) then kh, vhT
    _Float16* w16 = (_Float16*)d_ws;
    _Float16* kh  = w16 + 4 * 16384;                  // [64][224][128]
    _Float16* vhT = kh + (size_t)LWIN * KNP * DDIM;   // [64][128][224]

    hipFuncSetAttribute(reinterpret_cast<const void*>(kv_proj_kernel),
                        hipFuncAttributeMaxDynamicSharedMemorySize, 90112);
    hipFuncSetAttribute(reinterpret_cast<const void*>(attn_kernel),
                        hipFuncAttributeMaxDynamicSharedMemorySize, 155648);

    convert_w_kernel<<<64, 256, 0, stream>>>(Wq, Wk, Wv, Wp, w16);
    init_out_kernel<<<(out_size + 255) / 256, 256, 0, stream>>>(skip, out, out_size);
    kv_proj_kernel<<<LWIN, 256, 90112, stream>>>(k, v, gk, bkln, gv, bvln, bk, bv,
                                                 w16 + 16384, w16 + 32768, kh, vhT);
    attn_kernel<<<dim3(12, LWIN), 256, 155648, stream>>>(q, gq, bqln, bq, bp,
                                                         w16, w16 + 49152, kh, vhT, out);
}
